// GlobalAttentionLayer_44727789420892
// MI455X (gfx1250) — compile-verified, matrix operand order checked by reading, not run on hardware
//
#include <hip/hip_runtime.h>
#include <hip/hip_bf16.h>
#include <math.h>

// ---------------------------------------------------------------------------
// GlobalAttentionLayer for MI455X (gfx1250): bf16 WMMA GEMMs + fused epilogues
// B=8, N=8192, D=512, H=8, DH=64, WIDTH=2048, M = B*N = 65536 rows
// GEMM: 256 thr (8 waves), block tile 256x64, wave tile 32x64,
//       double-buffered LDS weight tiles via async global->LDS when available.
// ---------------------------------------------------------------------------

typedef unsigned short u16;
typedef unsigned int   u32;

typedef __attribute__((ext_vector_type(16))) __bf16 bf16x16;
typedef __attribute__((ext_vector_type(8)))  float  v8f;
typedef __attribute__((ext_vector_type(4)))  int    i32x4;

union FragB16 { bf16x16 v; uint4 q[2]; };

#if defined(__gfx1250__) && \
    __has_builtin(__builtin_amdgcn_global_load_async_to_lds_b128) && \
    __has_builtin(__builtin_amdgcn_s_wait_asynccnt)
#define USE_ASYNC_LDS 1
#define GLOBAL_AS __attribute__((address_space(1)))
#define LDS_AS    __attribute__((address_space(3)))
#else
#define USE_ASYNC_LDS 0
#endif

static __device__ __forceinline__ u16 f2bf(float f) {
    u32 u = __float_as_uint(f);
    u32 r = (u + 0x7FFFu + ((u >> 16) & 1u)) >> 16;   // round-to-nearest-even
    return (u16)r;
}
static __device__ __forceinline__ float bf2f(u16 h) {
    return __uint_as_float(((u32)h) << 16);
}
static __device__ __forceinline__ float gelu_exact(float v) {
    return 0.5f * v * (1.0f + erff(v * 0.70710678118654752f));
}

// ---------------------------------------------------------------------------
// fp32 -> bf16 elementwise (weight conversion)
// ---------------------------------------------------------------------------
__global__ void cvt_bf16_kernel(const float* __restrict__ in, u16* __restrict__ out, int n) {
    int i = blockIdx.x * 256 + threadIdx.x;
    if (i < n) out[i] = f2bf(in[i]);
}

// ---------------------------------------------------------------------------
// LayerNorm over D=512, optional row-mask multiply, bf16 output.
// One block (128 threads) per row.
// ---------------------------------------------------------------------------
__global__ void ln_kernel(const float* __restrict__ x,
                          const float* __restrict__ g,
                          const float* __restrict__ b,
                          const float* __restrict__ mask,
                          u16* __restrict__ out,
                          int applyMask) {
    const int D = 512;
    const int tid = threadIdx.x;
    const size_t row = blockIdx.x;
    const float* xr = x + row * D;

    float v4[4];
    float s = 0.f;
#pragma unroll
    for (int i = 0; i < 4; i++) { v4[i] = xr[tid + i * 128]; s += v4[i]; }

    __shared__ float red[128];
    red[tid] = s; __syncthreads();
    for (int off = 64; off > 0; off >>= 1) {
        if (tid < off) red[tid] += red[tid + off];
        __syncthreads();
    }
    float mean = red[0] * (1.0f / 512.0f);
    __syncthreads();

    float vs = 0.f;
#pragma unroll
    for (int i = 0; i < 4; i++) { float d = v4[i] - mean; vs += d * d; }
    red[tid] = vs; __syncthreads();
    for (int off = 64; off > 0; off >>= 1) {
        if (tid < off) red[tid] += red[tid + off];
        __syncthreads();
    }
    float rstd = rsqrtf(red[0] * (1.0f / 512.0f) + 1e-5f);
    float mw = applyMask ? mask[row] : 1.0f;

#pragma unroll
    for (int i = 0; i < 4; i++) {
        int col = tid + i * 128;
        float y = (v4[i] - mean) * rstd * g[col] + b[col];
        out[row * D + col] = f2bf(y * mw);
    }
}

// ---------------------------------------------------------------------------
// WMMA GEMM:  C[M,Nout] = A[M,K](bf16) * W[Nout,K](bf16)^T + bias
// Epilogue options: exact GELU, residual add (fp32 [M,Nout]), row mask, bf16 out.
// Block: 256 threads (8 waves); block tile 256x64; wave tile 32x64
// (2 A-fragments x 4 N-subtiles = 8 WMMA per k-step, 2 WMMAs per B-fragment).
// Weight k-tile (32k x 64n) double-buffered in LDS; next tile copied with
// GLOBAL_LOAD_ASYNC_TO_LDS_B128 (ASYNCcnt) while current tile is consumed.
// ---------------------------------------------------------------------------
template <bool GELU, bool RES, bool MASKED, bool OUTBF>
__global__ __launch_bounds__(256)
void gemm_wmma_kernel(const u16* __restrict__ A,
                      const u16* __restrict__ W,
                      const float* __restrict__ bias,
                      const float* __restrict__ residual,
                      const float* __restrict__ mask,
                      void* __restrict__ outp,
                      int M, int Nout, int K) {
    __shared__ u16 Bs[2][64 * 40];         // [buf][n][k], stride 40 (80B): no bank conflicts

    const int tid  = threadIdx.x;
    const int wave = tid >> 5;
    const int lane = tid & 31;
    const int lhalf = lane >> 4;           // selects K-half of fragment
    const int lrow  = lane & 15;           // A row / B column within tile

    const int n0 = blockIdx.x * 64;
    const int m0 = blockIdx.y * 256 + wave * 32;

    v8f acc[4][2] = {};                    // [n-subtile][m-subtile]

    const u16* Arow0 = A + (size_t)(m0 + lrow) * K;
    const u16* Arow1 = Arow0 + (size_t)16 * K;
    const int ldn = tid >> 2;              // 0..63 : n within tile
    const int ldk = (tid & 3) * 8;         // 0,8,16,24 : k within tile
    const u16* Wld = W + (size_t)(n0 + ldn) * K + ldk;
    u16* myLds0 = &Bs[0][ldn * 40 + ldk];
    u16* myLds1 = &Bs[1][ldn * 40 + ldk];

    // ---- prologue: stage k-tile 0 into buffer 0 ----
#if USE_ASYNC_LDS
    __builtin_amdgcn_global_load_async_to_lds_b128(
        (GLOBAL_AS i32x4*)Wld, (LDS_AS i32x4*)myLds0, 0, 0);
    __builtin_amdgcn_s_wait_asynccnt(0);
    __syncthreads();
#else
    *(uint4*)myLds0 = *(const uint4*)Wld;
    __syncthreads();
#endif

    int p = 0;
    for (int k0 = 0; k0 < K; k0 += 32) {
        const bool more = (k0 + 32) < K;

        // ---- issue next weight tile into the other buffer ----
#if USE_ASYNC_LDS
        if (more) {
            __builtin_amdgcn_global_load_async_to_lds_b128(
                (GLOBAL_AS i32x4*)(Wld + k0 + 32),
                (LDS_AS i32x4*)(p ? myLds0 : myLds1), 0, 0);
        }
#else
        uint4 stg;
        if (more) stg = *(const uint4*)(Wld + k0 + 32);
#endif
        if (more)                           // pull next A cachelines too
            __builtin_prefetch(Arow0 + k0 + 32, 0, 3);

        // ---- A fragments: 16x32 bf16, lane%16 = row, lane/16 = K-half ----
        FragB16 a0, a1;
        a0.q[0] = *(const uint4*)(Arow0 + k0 + lhalf * 8);
        a0.q[1] = *(const uint4*)(Arow0 + k0 + 16 + lhalf * 8);
        a1.q[0] = *(const uint4*)(Arow1 + k0 + lhalf * 8);
        a1.q[1] = *(const uint4*)(Arow1 + k0 + 16 + lhalf * 8);

        const u16* bufBase = &Bs[p][0];
#pragma unroll
        for (int j = 0; j < 4; j++) {
            FragB16 bfrag;                  // 32x16 bf16: lane%16 = col, K along regs
            const u16* bs = bufBase + (j * 16 + lrow) * 40;
            bfrag.q[0] = *(const uint4*)(bs + lhalf * 8);
            bfrag.q[1] = *(const uint4*)(bs + 16 + lhalf * 8);
            acc[j][0] = __builtin_amdgcn_wmma_f32_16x16x32_bf16(
                            false, a0.v, false, bfrag.v, (short)0, acc[j][0], false, false);
            acc[j][1] = __builtin_amdgcn_wmma_f32_16x16x32_bf16(
                            false, a1.v, false, bfrag.v, (short)0, acc[j][1], false, false);
        }

        // ---- retire the copy, swap buffers ----
        if (more) {
#if USE_ASYNC_LDS
            __builtin_amdgcn_s_wait_asynccnt(0);
            __syncthreads();
#else
            __syncthreads();
            *(uint4*)(p ? myLds0 : myLds1) = stg;
            __syncthreads();
#endif
            p ^= 1;
        }
    }

    // ---- epilogue: VGPR r -> M = base + r (lanes 0-15) / base+8+r (lanes 16-31)
#pragma unroll
    for (int j = 0; j < 4; j++) {
#pragma unroll
        for (int t = 0; t < 2; t++) {
#pragma unroll
            for (int r = 0; r < 8; r++) {
                int row = m0 + t * 16 + r + lhalf * 8;
                int col = n0 + j * 16 + lrow;
                float v = acc[j][t][r] + bias[col];
                if (GELU)   v = gelu_exact(v);
                if (RES)    v += residual[(size_t)row * Nout + col];
                if (MASKED) v *= mask[row];
                if (OUTBF)  ((u16*)outp)[(size_t)row * Nout + col] = f2bf(v);
                else        ((float*)outp)[(size_t)row * Nout + col] = v;
            }
        }
    }
}

// ---------------------------------------------------------------------------
// Stage 1 pooling: alpha = softmax_n(scale * Q.q_alpha); global_q = sum alpha*Q
// One block (256 threads) per (b,h). Q is fp32 [M, 512].
// ---------------------------------------------------------------------------
__global__ void pool1_kernel(const float* __restrict__ Qf,
                             const float* __restrict__ q_alpha,
                             const float* __restrict__ mask,
                             float* __restrict__ gq) {
    const int NN = 8192;
    const int tid = threadIdx.x;
    const int bh = blockIdx.x, b = bh >> 3, h = bh & 7;
    const float scale = 0.125f;                 // 64^-0.5

    __shared__ float sal[64], sacc[64], sred[256], sZ;
    if (tid < 64) { sal[tid] = q_alpha[h * 64 + tid]; sacc[tid] = 0.f; }
    if (tid == 0) sZ = 0.f;
    __syncthreads();

    float lmax = -1e30f;
    for (int n = tid; n < NN; n += 256) {
        if (mask[b * NN + n] != 0.f) {
            const float* q = Qf + ((size_t)(b * NN + n) * 512) + h * 64;
            float d = 0.f;
#pragma unroll
            for (int i = 0; i < 64; i++) d += q[i] * sal[i];
            lmax = fmaxf(lmax, d * scale);
        }
    }
    sred[tid] = lmax; __syncthreads();
    for (int off = 128; off > 0; off >>= 1) {
        if (tid < off) sred[tid] = fmaxf(sred[tid], sred[tid + off]);
        __syncthreads();
    }
    float gmax = sred[0];
    __syncthreads();

    float acc[64];
#pragma unroll
    for (int i = 0; i < 64; i++) acc[i] = 0.f;
    float z = 0.f;
    for (int n = tid; n < NN; n += 256) {
        if (mask[b * NN + n] != 0.f) {
            const float* q = Qf + ((size_t)(b * NN + n) * 512) + h * 64;
            float d = 0.f;
#pragma unroll
            for (int i = 0; i < 64; i++) d += q[i] * sal[i];
            float e = __expf(d * scale - gmax);
            z += e;
#pragma unroll
            for (int i = 0; i < 64; i++) acc[i] += e * q[i];
        }
    }
    atomicAdd(&sZ, z);                          // ds_add_f32
#pragma unroll
    for (int i = 0; i < 64; i++) atomicAdd(&sacc[i], acc[i]);
    __syncthreads();
    if (tid < 64) gq[bh * 64 + tid] = sacc[tid] / sZ;
}

// ---------------------------------------------------------------------------
// Stage 2 pooling: beta = softmax_n(scale * K.(global_q*k_beta)); gv = sum beta*V
// K,V are bf16 [M,512].
// ---------------------------------------------------------------------------
__global__ void pool2_kernel(const u16* __restrict__ Kb,
                             const u16* __restrict__ Vb,
                             const float* __restrict__ gq,
                             const float* __restrict__ k_beta,
                             const float* __restrict__ mask,
                             float* __restrict__ gv) {
    const int NN = 8192;
    const int tid = threadIdx.x;
    const int bh = blockIdx.x, b = bh >> 3, h = bh & 7;

    __shared__ float sw[64], sacc[64], sred[256], sZ;
    if (tid < 64) {
        sw[tid] = gq[bh * 64 + tid] * k_beta[h * 64 + tid] * 0.125f;
        sacc[tid] = 0.f;
    }
    if (tid == 0) sZ = 0.f;
    __syncthreads();

    float lmax = -1e30f;
    for (int n = tid; n < NN; n += 256) {
        if (mask[b * NN + n] != 0.f) {
            const u16* kr = Kb + ((size_t)(b * NN + n) * 512) + h * 64;
            float d = 0.f;
#pragma unroll
            for (int i = 0; i < 64; i++) d += bf2f(kr[i]) * sw[i];
            lmax = fmaxf(lmax, d);
        }
    }
    sred[tid] = lmax; __syncthreads();
    for (int off = 128; off > 0; off >>= 1) {
        if (tid < off) sred[tid] = fmaxf(sred[tid], sred[tid + off]);
        __syncthreads();
    }
    float gmax = sred[0];
    __syncthreads();

    float acc[64];
#pragma unroll
    for (int i = 0; i < 64; i++) acc[i] = 0.f;
    float z = 0.f;
    for (int n = tid; n < NN; n += 256) {
        if (mask[b * NN + n] != 0.f) {
            const u16* kr = Kb + ((size_t)(b * NN + n) * 512) + h * 64;
            const u16* vr = Vb + ((size_t)(b * NN + n) * 512) + h * 64;
            float d = 0.f;
#pragma unroll
            for (int i = 0; i < 64; i++) d += bf2f(kr[i]) * sw[i];
            float e = __expf(d - gmax);
            z += e;
#pragma unroll
            for (int i = 0; i < 64; i++) acc[i] += e * bf2f(vr[i]);
        }
    }
    atomicAdd(&sZ, z);
#pragma unroll
    for (int i = 0; i < 64; i++) atomicAdd(&sacc[i], acc[i]);
    __syncthreads();
    if (tid < 64) gv[bh * 64 + tid] = sacc[tid] / sZ;
}

// ---------------------------------------------------------------------------
// out_bf16[m, c] = Q[m, c] + global_v[b, c]   (c = h*64+d, b = m / 8192)
// ---------------------------------------------------------------------------
__global__ void add_gv_kernel(const float* __restrict__ Qf,
                              const float* __restrict__ gv,
                              u16* __restrict__ outbf) {
    size_t i = (size_t)blockIdx.x * 256 + threadIdx.x;
    int c = (int)(i & 511);
    size_t m = i >> 9;
    int b = (int)(m >> 13);                 // m / 8192
    outbf[i] = f2bf(Qf[i] + gv[(size_t)b * 512 + c]);
}

// ---------------------------------------------------------------------------
// Host-side orchestration
// ---------------------------------------------------------------------------
extern "C" void kernel_launch(void* const* d_in, const int* in_sizes, int n_in,
                              void* d_out, int out_size, void* d_ws, size_t ws_size,
                              hipStream_t stream) {
    const int B = 8, N = 8192, D = 512, WIDTH = 2048;
    const int M = B * N;                    // 65536

    const float* x      = (const float*)d_in[0];
    const float* mask   = (const float*)d_in[1];
    const float* Wq     = (const float*)d_in[2];
    const float* bq     = (const float*)d_in[3];
    const float* Wk     = (const float*)d_in[4];
    const float* bk     = (const float*)d_in[5];
    const float* Wv     = (const float*)d_in[6];
    const float* bv     = (const float*)d_in[7];
    const float* qalpha = (const float*)d_in[8];
    const float* kbeta  = (const float*)d_in[9];
    const float* Wo     = (const float*)d_in[10];
    const float* bo     = (const float*)d_in[11];
    const float* ln0g   = (const float*)d_in[12];
    const float* ln0b   = (const float*)d_in[13];
    const float* ln1g   = (const float*)d_in[14];
    const float* ln1b   = (const float*)d_in[15];
    const float* W1     = (const float*)d_in[16];
    const float* b1     = (const float*)d_in[17];
    const float* W2     = (const float*)d_in[18];
    const float* b2     = (const float*)d_in[19];

    // workspace layout (256B aligned, with aliasing)
    char* ws = (char*)d_ws;
    size_t off = 0;
    auto alloc = [&](size_t bytes) -> char* {
        char* p = ws + off;
        off += (bytes + 255) & ~(size_t)255;
        return p;
    };
    u16*   wq_bf = (u16*)alloc((size_t)D * D * 2);
    u16*   wk_bf = (u16*)alloc((size_t)D * D * 2);
    u16*   wv_bf = (u16*)alloc((size_t)D * D * 2);
    u16*   wo_bf = (u16*)alloc((size_t)D * D * 2);
    u16*   w1_bf = (u16*)alloc((size_t)WIDTH * D * 2);
    u16*   w2_bf = (u16*)alloc((size_t)D * WIDTH * 2);
    u16*   xn_bf = (u16*)alloc((size_t)M * D * 2);       // later reused as attn-out bf16
    float* q_f   = (float*)alloc((size_t)M * D * 4);     // later reused as x1 (post-attn)
    u16*   k_bf  = (u16*)alloc((size_t)M * D * 2);       // later reused as LN1-out bf16
    u16*   v_bf  = (u16*)alloc((size_t)M * D * 2);
    u16*   h1_bf = (u16*)alloc((size_t)M * WIDTH * 2);
    float* gq    = (float*)alloc((size_t)B * 8 * 64 * 4);
    float* gv    = (float*)alloc((size_t)B * 8 * 64 * 4);
    (void)ws_size; (void)n_in; (void)in_sizes; (void)out_size;

    // 1) convert weights to bf16
    cvt_bf16_kernel<<<(D * D + 255) / 256, 256, 0, stream>>>(Wq, wq_bf, D * D);
    cvt_bf16_kernel<<<(D * D + 255) / 256, 256, 0, stream>>>(Wk, wk_bf, D * D);
    cvt_bf16_kernel<<<(D * D + 255) / 256, 256, 0, stream>>>(Wv, wv_bf, D * D);
    cvt_bf16_kernel<<<(D * D + 255) / 256, 256, 0, stream>>>(Wo, wo_bf, D * D);
    cvt_bf16_kernel<<<(WIDTH * D + 255) / 256, 256, 0, stream>>>(W1, w1_bf, WIDTH * D);
    cvt_bf16_kernel<<<(WIDTH * D + 255) / 256, 256, 0, stream>>>(W2, w2_bf, WIDTH * D);

    // 2) LN0 * mask -> xn (bf16)
    ln_kernel<<<M, 128, 0, stream>>>(x, ln0g, ln0b, mask, xn_bf, 1);

    // 3) Q/K/V projections (WMMA bf16)
    dim3 gq512(D / 64, M / 256);
    gemm_wmma_kernel<false, false, false, false><<<gq512, 256, 0, stream>>>(
        xn_bf, wq_bf, bq, nullptr, nullptr, q_f, M, D, D);
    gemm_wmma_kernel<false, false, false, true><<<gq512, 256, 0, stream>>>(
        xn_bf, wk_bf, bk, nullptr, nullptr, k_bf, M, D, D);
    gemm_wmma_kernel<false, false, false, true><<<gq512, 256, 0, stream>>>(
        xn_bf, wv_bf, bv, nullptr, nullptr, v_bf, M, D, D);

    // 4) global pooling stages
    pool1_kernel<<<B * 8, 256, 0, stream>>>(q_f, qalpha, mask, gq);
    pool2_kernel<<<B * 8, 256, 0, stream>>>(k_bf, v_bf, gq, kbeta, mask, gv);

    // 5) attn_out = Q + broadcast(global_v)  -> bf16 (reuse xn_bf)
    add_gv_kernel<<<(int)(((size_t)M * D) / 256), 256, 0, stream>>>(q_f, gv, xn_bf);

    // 6) O-projection + residual(x) -> x1 fp32 (reuse q_f)
    gemm_wmma_kernel<false, true, false, false><<<gq512, 256, 0, stream>>>(
        xn_bf, wo_bf, bo, x, nullptr, q_f, M, D, D);

    // 7) LN1 -> bf16 (reuse k_bf)
    ln_kernel<<<M, 128, 0, stream>>>(q_f, ln1g, ln1b, nullptr, k_bf, 0);

    // 8) h = gelu(xn1 @ W1^T + b1) -> bf16 [M, 2048]
    dim3 g1(WIDTH / 64, M / 256);
    gemm_wmma_kernel<true, false, false, true><<<g1, 256, 0, stream>>>(
        k_bf, w1_bf, b1, nullptr, nullptr, h1_bf, M, WIDTH, D);

    // 9) out = (x1 + gelu(h @ W2^T + b2)) * mask -> fp32 d_out
    gemm_wmma_kernel<true, true, true, false><<<gq512, 256, 0, stream>>>(
        h1_bf, w2_bf, b2, q_f, mask, (float*)d_out, M, D, WIDTH);
}
